// HGCNN_geo_3513283248951
// MI455X (gfx1250) — matrix operand order repaired and checked by reading, not compiled, not run on hardware
//
#include <hip/hip_runtime.h>
#include <hip/hip_bf16.h>

// ---------------------------------------------------------------------------
// HGCNN_geo on MI455X (gfx1250, wave32, WMMA).
// Pipeline:
//   1) knn_kernel           : masked top-16 (local) / top-32 (global) per row
//   2) prep kernels         : f16 weight repack + BN fold (s,d)
//   3) centerT_kernel (WMMA): T = s*((wc-wd)·x) + d       per layer
//   4) edge_conv_kernel(WMMA): y = max_k lrelu(s*(wd·x_nbr) + T)
//   5) merge_xm             : per-point local/global select (512 ch)
//   6) final_gemm_kernel(WMMA): y9 = lrelu(s9*(w9·xm)+d9) + fused max/mean pool
//   7) head_kernel          : 3-layer MLP (tiny, scalar f32)
// f16 operands, f32 WMMA accumulation throughout the GEMMs.
// ---------------------------------------------------------------------------

#define B_  4
#define N_  2048
#define NEG_ 0.2f
#define EPS_ 1e-5f

typedef __attribute__((ext_vector_type(16))) _Float16 v16h;
typedef __attribute__((ext_vector_type(8)))  float    v8f;

__device__ __forceinline__ unsigned f2ord(float f) {
    unsigned u = __float_as_uint(f);
    return (u & 0x80000000u) ? ~u : (u | 0x80000000u);
}
__device__ __forceinline__ float ord2f(unsigned u) {
    unsigned v = (u & 0x80000000u) ? (u & 0x7fffffffu) : ~u;
    return __uint_as_float(v);
}
__device__ __forceinline__ float lrelu(float v) { return v < 0.f ? v * NEG_ : v; }

// ---------------------------------------------------------------------------
// KNN: reproduces  pd = where(allow, geod, rowmin); top_k(pd, k)
// One workgroup per (b, n) row; iterative argmax in LDS, ties -> lowest index.
// ---------------------------------------------------------------------------
__global__ __launch_bounds__(256) void knn_kernel(
    const float* __restrict__ geod, const int* __restrict__ lidx,
    int* __restrict__ idxL, int* __restrict__ idxG)
{
    __shared__ float raw[N_];
    __shared__ float pd[N_];
    __shared__ unsigned long long red[256];
    __shared__ float fred[256];
    __shared__ float smn;

    const int bn  = blockIdx.x;
    const int b   = bn / N_;
    const int tid = threadIdx.x;
    const float* row = geod + (size_t)bn * N_;

    float mn = 3.402823466e38f;
    for (int j = tid; j < N_; j += 256) { float v = row[j]; raw[j] = v; mn = fminf(mn, v); }
    fred[tid] = mn; __syncthreads();
    for (int s = 128; s > 0; s >>= 1) {
        if (tid < s) fred[tid] = fminf(fred[tid], fred[tid + s]);
        __syncthreads();
    }
    if (tid == 0) smn = fred[0];
    __syncthreads();

    const int* lrow = lidx + (size_t)b * N_;
    for (int pass = 0; pass < 2; ++pass) {
        const bool wantLocal = (pass == 0);
        const int  k   = wantLocal ? 16 : 32;
        int* outp = wantLocal ? (idxL + (size_t)bn * 16) : (idxG + (size_t)bn * 32);

        for (int j = tid; j < N_; j += 256) {
            bool allow = ((lrow[j] != 0) == wantLocal);
            pd[j] = allow ? raw[j] : smn;
        }
        __syncthreads();

        for (int it = 0; it < k; ++it) {
            unsigned long long best = 0ull;
            for (int j = tid; j < N_; j += 256) {
                unsigned long long key =
                    ((unsigned long long)f2ord(pd[j]) << 32) | (unsigned)(N_ - 1 - j);
                best = (key > best) ? key : best;
            }
            red[tid] = best; __syncthreads();
            for (int s = 128; s > 0; s >>= 1) {
                if (tid < s && red[tid + s] > red[tid]) red[tid] = red[tid + s];
                __syncthreads();
            }
            if (tid == 0) {
                int sel = (N_ - 1) - (int)(red[0] & 0xffffffffull);
                outp[it] = sel;
                pd[sel] = __uint_as_float(0xff800000u); // -inf: exclude
            }
            __syncthreads();
        }
    }
}

// ---------------------------------------------------------------------------
// Prep kernels: BN fold + f16 repack (zero-padded K dim for WMMA)
// ---------------------------------------------------------------------------
__global__ void prep_edge_w(const float* __restrict__ w, const float* __restrict__ bn,
                            int C, int Cpad, int O,
                            _Float16* __restrict__ Wd, _Float16* __restrict__ U,
                            float* __restrict__ s, float* __restrict__ d)
{
    int i = blockIdx.x * blockDim.x + threadIdx.x;
    if (i >= O * Cpad) return;
    int o = i / Cpad, c = i % Cpad;
    float wd = (c < C) ? w[(size_t)o * 2 * C + c]     : 0.f;
    float wc = (c < C) ? w[(size_t)o * 2 * C + C + c] : 0.f;
    Wd[i] = (_Float16)wd;
    U[i]  = (_Float16)(wc - wd);
    if (c == 0) {
        float g = bn[o], bb = bn[O + o], mm = bn[2 * O + o], vv = bn[3 * O + o];
        float sv = g * rsqrtf(vv + EPS_);
        s[o] = sv; d[o] = bb - mm * sv;
    }
}

__global__ void prep_x0_kernel(const float* __restrict__ x, _Float16* __restrict__ X0)
{
    int i = blockIdx.x * blockDim.x + threadIdx.x;
    if (i >= B_ * N_ * 32) return;
    int c = i & 31;
    int n = (i >> 5) % N_;
    int b = i / (32 * N_);
    X0[i] = (c < 3) ? (_Float16)x[((size_t)b * 3 + c) * N_ + n] : (_Float16)0.f;
}

__global__ void prep_w9_kernel(const float* __restrict__ w9, const float* __restrict__ bn9,
                               _Float16* __restrict__ W9h,
                               float* __restrict__ s9, float* __restrict__ d9)
{
    int i = blockIdx.x * blockDim.x + threadIdx.x;
    if (i < 1024 * 512) W9h[i] = (_Float16)w9[i];
    if (i < 1024) {
        float g = bn9[i], bb = bn9[1024 + i], mm = bn9[2048 + i], vv = bn9[3072 + i];
        float sv = g * rsqrtf(vv + EPS_);
        s9[i] = sv; d9[i] = bb - mm * sv;
    }
}

__global__ void init_pool_kernel(unsigned* __restrict__ pmax, float* __restrict__ psum)
{
    int i = blockIdx.x * blockDim.x + threadIdx.x;
    if (i < B_ * 1024) { pmax[i] = 0u; psum[i] = 0.f; }
}

// ---------------------------------------------------------------------------
// WMMA fragment loaders (ISA 7.12.2 layouts, wave32)
//   A 16x32 f16 : row M = lane%16; K-halves split 0-7/16-23 vs 8-15/24-31
//   B 32x16 f16 : col N = lane%16; K-halves split 0-7/16-23 vs 8-15/24-31 (same as A)
//   C/D 16x16 f32: col = lane%16, row = vgpr + 8*(lane>>4)
// ---------------------------------------------------------------------------
__device__ __forceinline__ v16h load_a_frag(const _Float16* __restrict__ rowp, int hs)
{
    v16h a;
    const _Float16* ar = rowp + 8 * hs;
    #pragma unroll
    for (int j = 0; j < 8; ++j) { a[j] = ar[j]; a[8 + j] = ar[16 + j]; }
    return a;
}
__device__ __forceinline__ v16h load_b_frag(const _Float16* __restrict__ colp, int hs)
{
    v16h bf;
    const _Float16* br = colp + 8 * hs;
    #pragma unroll
    for (int j = 0; j < 8; ++j) { bf[j] = br[j]; bf[8 + j] = br[16 + j]; }
    return bf;
}

// ---------------------------------------------------------------------------
// Center-term GEMM: T[b,n,o] = s[o] * sum_c U[o,c]*X[b,n,c] + d[o]
// One workgroup per 16-point group; one wave per 16-row O tile.
// ---------------------------------------------------------------------------
template<int CPAD, int O>
__global__ __launch_bounds__((O / 16) * 32) void centerT_kernel(
    const _Float16* __restrict__ Xin, int instride, int inoff,
    const _Float16* __restrict__ U,
    const float* __restrict__ s, const float* __restrict__ dsh,
    float* __restrict__ T)
{
    constexpr int NW = O / 16;
    constexpr int BD = NW * 32;
    __shared__ __align__(16) _Float16 Bl[16 * CPAD];

    const int blk = blockIdx.x;
    const int b   = blk / (N_ / 16);
    const int n0  = (blk % (N_ / 16)) * 16;
    const int tid = threadIdx.x;

    constexpr int C2 = CPAD / 2;
    for (int i = tid; i < 16 * C2; i += BD) {
        int j = i / C2, c2 = i % C2;
        const unsigned* src =
            (const unsigned*)(Xin + (size_t)(b * N_ + n0 + j) * instride + inoff);
        ((unsigned*)Bl)[i] = src[c2];
    }
    __syncthreads();

    const int lane = tid & 31, m = lane & 15, hs = lane >> 4;
    const int om = (tid >> 5) * 16;

    v8f acc = {};
    for (int kk = 0; kk < CPAD; kk += 32) {
        v16h a  = load_a_frag(U + (size_t)(om + m) * CPAD + kk, hs);
        v16h bf = load_b_frag(Bl + m * CPAD + kk, hs);
        acc = __builtin_amdgcn_wmma_f32_16x16x32_f16(false, a, false, bf,
                                                     (short)0, acc, false, false);
    }

    float* tp = T + (size_t)(b * N_ + n0 + m) * O + om + 8 * hs;
    #pragma unroll
    for (int r = 0; r < 8; ++r) {
        int o = om + 8 * hs + r;
        tp[r] = s[o] * acc[r] + dsh[o];
    }
}

// ---------------------------------------------------------------------------
// EdgeConv GEMM: one workgroup per point. Columns of the B tile are the k
// gathered neighbors; max-over-k is a 16-lane shuffle reduce on the C frag.
// ---------------------------------------------------------------------------
template<int CPAD, int O, int K>
__global__ __launch_bounds__((O / 16) * 32) void edge_conv_kernel(
    const _Float16* __restrict__ Xin, int instride, int inoff,
    const int* __restrict__ idx,
    const _Float16* __restrict__ Wd, const float* __restrict__ s,
    const float* __restrict__ T,
    _Float16* __restrict__ Out, int outstride, int outoff)
{
    constexpr int NW = O / 16;
    constexpr int BD = NW * 32;
    __shared__ __align__(16) _Float16 Bl[K * CPAD];
    __shared__ int nidx[K];

    const int bn  = blockIdx.x;
    const int b   = bn / N_;
    const int tid = threadIdx.x;

    const int* ip = idx + (size_t)bn * K;
    if (tid < K) nidx[tid] = ip[tid];
    __syncthreads();

    constexpr int C2 = CPAD / 2;
    for (int i = tid; i < K * C2; i += BD) {
        int j = i / C2, c2 = i % C2;
        const unsigned* src =
            (const unsigned*)(Xin + (size_t)(b * N_ + nidx[j]) * instride + inoff);
        ((unsigned*)Bl)[i] = src[c2];
    }
    __syncthreads();

    const int lane = tid & 31, m = lane & 15, hs = lane >> 4;
    const int om = (tid >> 5) * 16;

    v8f acc0 = {};
    v8f acc1 = {};
    for (int kk = 0; kk < CPAD; kk += 32) {
        v16h a  = load_a_frag(Wd + (size_t)(om + m) * CPAD + kk, hs);
        v16h b0 = load_b_frag(Bl + m * CPAD + kk, hs);
        acc0 = __builtin_amdgcn_wmma_f32_16x16x32_f16(false, a, false, b0,
                                                      (short)0, acc0, false, false);
        if constexpr (K == 32) {
            v16h b1 = load_b_frag(Bl + (16 + m) * CPAD + kk, hs);
            acc1 = __builtin_amdgcn_wmma_f32_16x16x32_f16(false, a, false, b1,
                                                          (short)0, acc1, false, false);
        }
    }

    const float* tp = T + (size_t)bn * O + om + 8 * hs;
    float vr[8];
    #pragma unroll
    for (int r = 0; r < 8; ++r) {
        float sv = s[om + 8 * hs + r];
        float t  = tp[r];
        float v  = lrelu(sv * acc0[r] + t);
        if constexpr (K == 32) {
            float v1 = lrelu(sv * acc1[r] + t);
            v = fmaxf(v, v1);
        }
        vr[r] = v;
    }
    #pragma unroll
    for (int off = 8; off > 0; off >>= 1) {
        #pragma unroll
        for (int r = 0; r < 8; ++r)
            vr[r] = fmaxf(vr[r], __shfl_xor(vr[r], off, 32));
    }
    if (m == 0) {
        _Float16* op = Out + (size_t)bn * outstride + outoff + om + 8 * hs;
        #pragma unroll
        for (int r = 0; r < 8; ++r) op[r] = (_Float16)vr[r];
    }
}

// ---------------------------------------------------------------------------
// Per-point local/global select: Xm[b,n,:] = local ? CatL : CatG (512 ch)
// ---------------------------------------------------------------------------
__global__ void merge_xm_kernel(const int* __restrict__ lidx,
                                const _Float16* __restrict__ CatL,
                                const _Float16* __restrict__ CatG,
                                _Float16* __restrict__ Xm)
{
    int i = blockIdx.x * blockDim.x + threadIdx.x;
    if (i >= B_ * N_ * 256) return;
    int pnt = i >> 8;
    const unsigned* src = (const unsigned*)((lidx[pnt] != 0) ? CatL : CatG);
    ((unsigned*)Xm)[i] = src[i];
}

// ---------------------------------------------------------------------------
// Final 1024x512 GEMM + BN + lrelu + fused max/mean pooling (atomics).
// One workgroup per 16 points; 16 waves x 4 O-tiles each.
// ---------------------------------------------------------------------------
__global__ __launch_bounds__(512) void final_gemm_kernel(
    const _Float16* __restrict__ Xm, const _Float16* __restrict__ W9,
    const float* __restrict__ s9, const float* __restrict__ d9,
    unsigned* __restrict__ pmax, float* __restrict__ psum)
{
    __shared__ __align__(16) _Float16 Bl[16 * 512];

    const int blk = blockIdx.x;
    const int b   = blk / (N_ / 16);
    const int n0  = (blk % (N_ / 16)) * 16;
    const int tid = threadIdx.x;

    const unsigned* src = (const unsigned*)Xm;
    for (int i = tid; i < 16 * 256; i += 512) {
        int j = i >> 8, c2 = i & 255;
        ((unsigned*)Bl)[i] = src[(size_t)(b * N_ + n0 + j) * 256 + c2];
    }
    __syncthreads();

    const int wave = tid >> 5, lane = tid & 31, m = lane & 15, hs = lane >> 4;

    v8f acc[4];
    v8f zero = {};
    #pragma unroll
    for (int i = 0; i < 4; ++i) acc[i] = zero;

    for (int kk = 0; kk < 512; kk += 32) {
        v16h bf = load_b_frag(Bl + m * 512 + kk, hs);
        #pragma unroll
        for (int i = 0; i < 4; ++i) {
            int ot = wave * 4 + i;
            v16h a = load_a_frag(W9 + (size_t)(ot * 16 + m) * 512 + kk, hs);
            acc[i] = __builtin_amdgcn_wmma_f32_16x16x32_f16(false, a, false, bf,
                                                            (short)0, acc[i], false, false);
        }
    }

    #pragma unroll
    for (int i = 0; i < 4; ++i) {
        int ot = wave * 4 + i;
        float vm[8], vs[8];
        #pragma unroll
        for (int r = 0; r < 8; ++r) {
            int o = ot * 16 + 8 * hs + r;
            float v = lrelu(s9[o] * acc[i][r] + d9[o]);
            vm[r] = v; vs[r] = v;
        }
        #pragma unroll
        for (int off = 8; off > 0; off >>= 1) {
            #pragma unroll
            for (int r = 0; r < 8; ++r) {
                vm[r] = fmaxf(vm[r], __shfl_xor(vm[r], off, 32));
                vs[r] = vs[r] + __shfl_xor(vs[r], off, 32);
            }
        }
        if (m == 0) {
            #pragma unroll
            for (int r = 0; r < 8; ++r) {
                int o = ot * 16 + 8 * hs + r;
                atomicMax(pmax + (size_t)b * 1024 + o, f2ord(vm[r]));
                atomicAdd(psum + (size_t)b * 1024 + o, vs[r]);
            }
        }
    }
}

// ---------------------------------------------------------------------------
// MLP head (tiny): p = [max||mean] -> lin1/bn10/lrelu -> lin2/bn11/lrelu -> lin3
// ---------------------------------------------------------------------------
__global__ __launch_bounds__(256) void head_kernel(
    const unsigned* __restrict__ pmax, const float* __restrict__ psum,
    const float* __restrict__ lin1, const float* __restrict__ bn10,
    const float* __restrict__ lin2, const float* __restrict__ lin2b,
    const float* __restrict__ bn11,
    const float* __restrict__ lin3, const float* __restrict__ lin3b,
    float* __restrict__ out)
{
    __shared__ float p[2048];
    __shared__ float h1[512];
    __shared__ float h2[256];
    const int b = blockIdx.x, tid = threadIdx.x;

    for (int i = tid; i < 1024; i += 256) {
        p[i]        = ord2f(pmax[b * 1024 + i]);
        p[1024 + i] = psum[b * 1024 + i] * (1.f / 2048.f);
    }
    __syncthreads();

    for (int j = tid; j < 512; j += 256) {
        float acc = 0.f;
        const float* wr = lin1 + (size_t)j * 2048;
        for (int c = 0; c < 2048; ++c) acc += p[c] * wr[c];
        float g = bn10[j], bb = bn10[512 + j], mm = bn10[1024 + j], vv = bn10[1536 + j];
        h1[j] = lrelu((acc - mm) * (g * rsqrtf(vv + EPS_)) + bb);
    }
    __syncthreads();

    if (tid < 256) {
        int j = tid;
        float acc = lin2b[j];
        const float* wr = lin2 + (size_t)j * 512;
        for (int c = 0; c < 512; ++c) acc += h1[c] * wr[c];
        float g = bn11[j], bb = bn11[256 + j], mm = bn11[512 + j], vv = bn11[768 + j];
        h2[j] = lrelu((acc - mm) * (g * rsqrtf(vv + EPS_)) + bb);
    }
    __syncthreads();

    if (tid < 40) {
        float acc = lin3b[tid];
        const float* wr = lin3 + (size_t)tid * 256;
        for (int c = 0; c < 256; ++c) acc += h2[c] * wr[c];
        out[b * 40 + tid] = acc;
    }
}

// ---------------------------------------------------------------------------
// Host-side layer driver
// ---------------------------------------------------------------------------
template<int CPAD, int O, int K>
static inline void run_edge_layer(const float* w, const float* bn, int Cin,
                                  const _Float16* Xin, int instride, int inoff,
                                  const int* idx, _Float16* Out, int outoff,
                                  _Float16* Wd, _Float16* U, float* s, float* d,
                                  float* T, hipStream_t stream)
{
    int tot = O * CPAD;
    prep_edge_w<<<(tot + 255) / 256, 256, 0, stream>>>(w, bn, Cin, CPAD, O, Wd, U, s, d);
    centerT_kernel<CPAD, O><<<(B_ * N_) / 16, (O / 16) * 32, 0, stream>>>(
        Xin, instride, inoff, U, s, d, T);
    edge_conv_kernel<CPAD, O, K><<<B_ * N_, (O / 16) * 32, 0, stream>>>(
        Xin, instride, inoff, idx, Wd, s, T, Out, 512, outoff);
}

extern "C" void kernel_launch(void* const* d_in, const int* in_sizes, int n_in,
                              void* d_out, int out_size, void* d_ws, size_t ws_size,
                              hipStream_t stream)
{
    (void)in_sizes; (void)n_in; (void)out_size; (void)ws_size;

    const float* x    = (const float*)d_in[0];
    const int*   lidx = (const int*)d_in[1];
    const float* geod = (const float*)d_in[2];
    const float* w[9];
    for (int i = 0; i < 9; ++i)  w[i]   = (const float*)d_in[3 + i];
    const float* bnp[11];
    for (int i = 0; i < 11; ++i) bnp[i] = (const float*)d_in[12 + i];
    const float* lin1  = (const float*)d_in[23];
    const float* lin2  = (const float*)d_in[24];
    const float* lin2b = (const float*)d_in[25];
    const float* lin3  = (const float*)d_in[26];
    const float* lin3b = (const float*)d_in[27];
    float* out = (float*)d_out;

    char*  base = (char*)d_ws;
    size_t off  = 0;
    auto alloc = [&](size_t bytes) -> char* {
        char* p = base + off;
        off = (off + bytes + 255) & ~(size_t)255;
        return p;
    };
    int*      idxL = (int*)alloc((size_t)B_ * N_ * 16 * 4);
    int*      idxG = (int*)alloc((size_t)B_ * N_ * 32 * 4);
    _Float16* X0h  = (_Float16*)alloc((size_t)B_ * N_ * 32 * 2);
    _Float16* CatL = (_Float16*)alloc((size_t)B_ * N_ * 512 * 2);
    _Float16* CatG = (_Float16*)alloc((size_t)B_ * N_ * 512 * 2);
    _Float16* Xm   = (_Float16*)alloc((size_t)B_ * N_ * 512 * 2);
    float*    T    = (float*)alloc((size_t)B_ * N_ * 256 * 4);
    _Float16* Wd   = (_Float16*)alloc((size_t)256 * 128 * 2);
    _Float16* U    = (_Float16*)alloc((size_t)256 * 128 * 2);
    float*    sbuf = (float*)alloc(1024 * 4);
    float*    dbuf = (float*)alloc(1024 * 4);
    _Float16* W9h  = (_Float16*)alloc((size_t)1024 * 512 * 2);
    unsigned* pmax = (unsigned*)alloc(4096 * 4);
    float*    psum = (float*)alloc(4096 * 4);

    // 1) KNN (both masks)
    knn_kernel<<<B_ * N_, 256, 0, stream>>>(geod, lidx, idxL, idxG);

    // 2) x -> f16 point-major, zero-padded to 32 channels
    {
        int tot = B_ * N_ * 32;
        prep_x0_kernel<<<(tot + 255) / 256, 256, 0, stream>>>(x, X0h);
    }

    // 3) local chain (k=16) -> CatL channels [0,64,128,256)
    run_edge_layer< 32,  64, 16>(w[0], bnp[0],   3, X0h,  32,   0, idxL, CatL,   0, Wd, U, sbuf, dbuf, T, stream);
    run_edge_layer< 64,  64, 16>(w[1], bnp[1],  64, CatL, 512,  0, idxL, CatL,  64, Wd, U, sbuf, dbuf, T, stream);
    run_edge_layer< 64, 128, 16>(w[2], bnp[2],  64, CatL, 512, 64, idxL, CatL, 128, Wd, U, sbuf, dbuf, T, stream);
    run_edge_layer<128, 256, 16>(w[3], bnp[3], 128, CatL, 512,128, idxL, CatL, 256, Wd, U, sbuf, dbuf, T, stream);

    // 4) global chain (k=32) -> CatG
    run_edge_layer< 32,  64, 32>(w[4], bnp[4],   3, X0h,  32,   0, idxG, CatG,   0, Wd, U, sbuf, dbuf, T, stream);
    run_edge_layer< 64,  64, 32>(w[5], bnp[5],  64, CatG, 512,  0, idxG, CatG,  64, Wd, U, sbuf, dbuf, T, stream);
    run_edge_layer< 64, 128, 32>(w[6], bnp[6],  64, CatG, 512, 64, idxG, CatG, 128, Wd, U, sbuf, dbuf, T, stream);
    run_edge_layer<128, 256, 32>(w[7], bnp[7], 128, CatG, 512,128, idxG, CatG, 256, Wd, U, sbuf, dbuf, T, stream);

    // 5) per-point select
    {
        int tot = B_ * N_ * 256;
        merge_xm_kernel<<<(tot + 255) / 256, 256, 0, stream>>>(lidx, CatL, CatG, Xm);
    }

    // 6) final GEMM + fused pooling
    {
        int tot = 1024 * 512;
        prep_w9_kernel<<<(tot + 255) / 256, 256, 0, stream>>>(w[8], bnp[8], W9h, sbuf, dbuf);
        init_pool_kernel<<<(B_ * 1024 + 255) / 256, 256, 0, stream>>>(pmax, psum);
        final_gemm_kernel<<<B_ * (N_ / 16), 512, 0, stream>>>(Xm, W9h, sbuf, dbuf, pmax, psum);
    }

    // 7) MLP head
    head_kernel<<<B_, 256, 0, stream>>>(pmax, psum, lin1, bnp[9], lin2, lin2b,
                                        bnp[10], lin3, lin3b, out);
}
